// Attention_90855738180095
// MI455X (gfx1250) — matrix operand order repaired and checked by reading, not compiled, not run on hardware
//
#include <hip/hip_runtime.h>
#include <hip/hip_bf16.h>

// ---------------------------------------------------------------------------
// Types for CDNA5 WMMA (wave32): D(16x16 f32) = A(16x32 bf16) * B(32x16 bf16) + C
// ---------------------------------------------------------------------------
typedef __attribute__((ext_vector_type(16))) __bf16 v16bf;
typedef __attribute__((ext_vector_type(8)))  float  v8f;

struct alignas(16) U4 { unsigned x, y, z, w; };
struct alignas(8)  U2 { unsigned x, y; };

union Frag {
  U4    q[2];   // 32 bytes = 16 bf16 per lane
  v16bf v;
};

__device__ __forceinline__ unsigned short f2bf(float f) {
  unsigned u = __float_as_uint(f);
  u += 0x7FFFu + ((u >> 16) & 1u);          // round-to-nearest-even
  return (unsigned short)(u >> 16);
}

// Async global->LDS copy, 16 bytes per lane (GLOBAL_LOAD_ASYNC_TO_LDS_B128,
// GVS mode: SGPR64 base + signed VGPR32 byte offset).  VDST VGPR carries the
// LDS byte address (low 32 bits of the generic pointer per aperture rules).
// Tracked on ASYNCcnt; pair with s_wait_asynccnt before consuming.
__device__ __forceinline__ void async_copy_b128(const void* gbase,
                                                unsigned voff_bytes,
                                                void* lds_dst) {
  const unsigned lds_addr = (unsigned)(size_t)lds_dst;
  asm volatile("global_load_async_to_lds_b128 %0, %1, %2"
               :
               : "v"(lds_addr), "v"(voff_bytes), "s"(gbase)
               : "memory");
}

__device__ __forceinline__ void wait_asynccnt0() {
  asm volatile("s_wait_asynccnt 0x0" ::: "memory");
}

// A-matrix fragment (16x32, bf16), source row-major [m][k] with leading dim ld.
// ISA layout: lanes 0-15 (M=lane): K {0..7} in VGPR0-3, K {16..23} in VGPR4-7;
//             lanes 16-31 (M=lane-16): K {8..15}, K {24..31}.
__device__ __forceinline__ Frag load_afrag(const unsigned short* base, int m,
                                           int ld, int k0, int hf) {
  Frag f;
  const unsigned short* r = base + m * ld + k0 + hf * 8;
  f.q[0] = *(const U4*)(r);
  f.q[1] = *(const U4*)(r + 16);
  return f;
}

// B-matrix fragment (32x16, bf16), source is B^T row-major: Bt[n][k], ld elems.
// ISA layout (matches A): lanes 0-15 (N=lane): K {0..7}, K {16..23};
// lanes 16-31 (N=lane-16): K {8..15}, K {24..31}; 2 values per VGPR ascending.
__device__ __forceinline__ Frag load_bfrag(const unsigned short* base, int row,
                                           int ld, int k0, int hf) {
  Frag f;
  const unsigned short* r = base + row * ld + k0 + hf * 8;
  f.q[0] = *(const U4*)(r);
  f.q[1] = *(const U4*)(r + 16);
  return f;
}

__device__ __forceinline__ v8f wmma_bf16(const Frag& a, const Frag& b, v8f c) {
  return __builtin_amdgcn_wmma_f32_16x16x32_bf16(false, a.v, false, b.v,
                                                 (short)0, c, false, false);
}

#define V8F_ZERO {0.f, 0.f, 0.f, 0.f, 0.f, 0.f, 0.f, 0.f}

// ---------------------------------------------------------------------------
// Kernel 1: QKV GEMM.  Y[65536, 2304] = X[65536, 768] @ Wqkv + b.
// Output scattered to Q[bh][n][64], K[bh][n][64], Vt[bh][64][n] in bf16.
// Tile: 128(M) x 64(N), 8 waves, 24 K-steps of 32.
// ---------------------------------------------------------------------------
__global__ __launch_bounds__(256) void qkv_gemm_kernel(
    const float* __restrict__ x, const float* __restrict__ w,
    const float* __restrict__ bias,
    unsigned short* __restrict__ Qb, unsigned short* __restrict__ Kb,
    unsigned short* __restrict__ Vt) {
  __shared__ unsigned short As[128 * 32];   // A tile, row-major bf16
  __shared__ unsigned short Bt[64 * 32];    // B tile transposed: Bt[n][k]

  const int tid  = threadIdx.x;
  const int wv   = tid >> 5;
  const int lane = tid & 31;
  const int hf   = lane >> 4;
  const int nn   = lane & 15;

  const int m0    = blockIdx.y * 128;
  const int nt    = blockIdx.x;       // 0..35; 64-wide tile -> one (three, head)
  const int n0    = nt * 64;
  const int three = nt / 12;
  const int head  = nt % 12;

  v8f acc[4] = {V8F_ZERO, V8F_ZERO, V8F_ZERO, V8F_ZERO};

  for (int ks = 0; ks < 24; ++ks) {
    const int k0 = ks * 32;
    // Stage A: 128x32 fp32 -> bf16 (1024 float4, 4 per thread) and prefetch
    // the next K-step's A rows into cache while this step computes.
#pragma unroll
    for (int i = 0; i < 4; ++i) {
      const int idx = tid + i * 256;
      const int row = idx >> 3, c4 = idx & 7;
      const float* src = x + (size_t)(m0 + row) * 768 + k0 + c4 * 4;
      const float4 vv = *(const float4*)(src);
      __builtin_prefetch(src + 32, 0, 1);   // next K-step, same row
      U2 p;
      p.x = (unsigned)f2bf(vv.x) | ((unsigned)f2bf(vv.y) << 16);
      p.y = (unsigned)f2bf(vv.z) | ((unsigned)f2bf(vv.w) << 16);
      *(U2*)(As + row * 32 + c4 * 4) = p;
    }
    // Stage B transposed: W[k0+kk][n0+nn] -> Bt[nn][kk] (512 float4, 2/thread)
#pragma unroll
    for (int i = 0; i < 2; ++i) {
      const int idx = tid + i * 256;
      const int kk = idx >> 4, n4 = idx & 15;
      const float4 vv =
          *(const float4*)(w + (size_t)(k0 + kk) * 2304 + n0 + n4 * 4);
      Bt[(n4 * 4 + 0) * 32 + kk] = f2bf(vv.x);
      Bt[(n4 * 4 + 1) * 32 + kk] = f2bf(vv.y);
      Bt[(n4 * 4 + 2) * 32 + kk] = f2bf(vv.z);
      Bt[(n4 * 4 + 3) * 32 + kk] = f2bf(vv.w);
    }
    __syncthreads();
    const Frag a = load_afrag(As, wv * 16 + nn, 32, 0, hf);
#pragma unroll
    for (int j = 0; j < 4; ++j) {
      const Frag b = load_bfrag(Bt, j * 16 + nn, 32, 0, hf);
      acc[j] = wmma_bf16(a, b, acc[j]);
    }
    __syncthreads();
  }

  // Epilogue: add bias, convert, scatter to Q / K / Vt.
#pragma unroll
  for (int j = 0; j < 4; ++j) {
    const int n_out = n0 + j * 16 + nn;
    const float bv  = bias[n_out];
    const int ch    = n_out & 63;
#pragma unroll
    for (int r = 0; r < 8; ++r) {
      const int tok  = m0 + wv * 16 + r + hf * 8;
      const int bidx = tok >> 8;
      const int ntok = tok & 255;
      const unsigned short o = f2bf(acc[j][r] + bv);
      if (three == 0)
        Qb[(((size_t)bidx * 12 + head) * 256 + ntok) * 64 + ch] = o;
      else if (three == 1)
        Kb[(((size_t)bidx * 12 + head) * 256 + ntok) * 64 + ch] = o;
      else
        Vt[(((size_t)bidx * 12 + head) * 64 + ch) * 256 + ntok] = o;
    }
  }
}

// ---------------------------------------------------------------------------
// Kernel 2: attention per (batch, head).  Block = 256 threads (8 waves).
// Each wave processes two 16-query row blocks (q_h constant per block).
// K and V^T staged via async global->LDS DMA (ASYNCcnt).
// ---------------------------------------------------------------------------
__global__ __launch_bounds__(256) void attn_kernel(
    const unsigned short* __restrict__ Qb, const unsigned short* __restrict__ Kb,
    const unsigned short* __restrict__ Vt,
    const float* __restrict__ rel_pos_h, const float* __restrict__ rel_pos_w,
    unsigned short* __restrict__ Ob) {
  extern __shared__ unsigned short smem[];
  unsigned short* Ksm = smem;                 // 256*64  K row-major [key][d]
  unsigned short* Vsm = Ksm + 256 * 64;       // 64*256  V^T [ch][key]
  unsigned short* Rh  = Vsm + 64 * 256;       // 32*64   rel_pos_h bf16 (padded)
  unsigned short* Rw  = Rh + 32 * 64;         // 32*64   rel_pos_w bf16
  unsigned short* Pm  = Rw + 32 * 64;         // 8 waves * 16*256 probs (bf16)

  const int tid  = threadIdx.x;
  const int wv   = tid >> 5;
  const int lane = tid & 31;
  const int hf   = lane >> 4;
  const int nn   = lane & 15;
  const int bh   = blockIdx.x;
  const int head = bh % 12, bidx = bh / 12;

  const unsigned short* Qp = Qb + (size_t)bh * 256 * 64;
  const unsigned short* Kp = Kb + (size_t)bh * 256 * 64;
  const unsigned short* Vp = Vt + (size_t)bh * 64 * 256;

  // Stage K and V^T into LDS with async b128 DMA (2048 x 16B each).
#pragma unroll
  for (int i = 0; i < 8; ++i) {
    const int idx = tid + i * 256;
    async_copy_b128(Kp, (unsigned)idx * 16u, Ksm + idx * 8);
    async_copy_b128(Vp, (unsigned)idx * 16u, Vsm + idx * 8);
  }
  // Rel-pos tables fp32 -> bf16 (31*64 = 1984 each) via VALU.
#pragma unroll
  for (int i = 0; i < 8; ++i) {
    const int idx = tid + i * 256;
    if (idx < 31 * 64) {
      Rh[idx] = f2bf(rel_pos_h[idx]);
      Rw[idx] = f2bf(rel_pos_w[idx]);
    }
  }
  wait_asynccnt0();          // barrier does NOT imply an ASYNCcnt wait
  __syncthreads();

  unsigned short* Pw = Pm + wv * (16 * 256);  // this wave's prob tile

  for (int it = 0; it < 2; ++it) {
    const int rblk = wv * 2 + it;   // 16-query row block; q_h == rblk
    const int qh   = rblk;

    const Frag aq0 = load_afrag(Qp, rblk * 16 + nn, 64, 0,  hf);
    const Frag aq1 = load_afrag(Qp, rblk * 16 + nn, 64, 32, hf);

    // rel_h tile (16x16): B rows indexed by dist = qh - kh + 15 (kh = lane N).
    v8f hacc = V8F_ZERO;
    {
      const int dist = qh - nn + 15;
      const Frag b0 = load_bfrag(Rh, dist, 64, 0,  hf);
      const Frag b1 = load_bfrag(Rh, dist, 64, 32, hf);
      hacc = wmma_bf16(aq0, b0, hacc);
      hacc = wmma_bf16(aq1, b1, hacc);
    }
    // rel_w: U[mm][dd] = q[mm] . rel_pos_w[dd], dd in [0,31) (two 16-col tiles)
    v8f u0 = V8F_ZERO, u1 = V8F_ZERO;
    {
      const Frag b0 = load_bfrag(Rw, nn, 64, 0,  hf);
      const Frag b1 = load_bfrag(Rw, nn, 64, 32, hf);
      u0 = wmma_bf16(aq0, b0, u0);
      u0 = wmma_bf16(aq1, b1, u0);
      const Frag c0 = load_bfrag(Rw, 16 + nn, 64, 0,  hf);
      const Frag c1 = load_bfrag(Rw, 16 + nn, 64, 32, hf);
      u1 = wmma_bf16(aq0, c0, u1);
      u1 = wmma_bf16(aq1, c1, u1);
    }

    // Scores: 16 column tiles; bias gathered via ds_bpermute from C-layout.
    float s[16][8];
#pragma unroll
    for (int j = 0; j < 16; ++j) {
      v8f sa = V8F_ZERO;
      const Frag b0 = load_bfrag(Ksm, j * 16 + nn, 64, 0,  hf);
      const Frag b1 = load_bfrag(Ksm, j * 16 + nn, 64, 32, hf);
      sa = wmma_bf16(aq0, b0, sa);
      sa = wmma_bf16(aq1, b1, sa);
#pragma unroll
      for (int r = 0; r < 8; ++r) {
        const float bias_h = __shfl(hacc[r], j + (lane & 16), 32);
        const int mm   = r + hf * 8;
        const int col  = mm - nn + 15;            // 0..30
        const int srcl = (col & 15) + (lane & 16);
        const float w0 = __shfl(u0[r], srcl, 32);
        const float w1 = __shfl(u1[r], srcl, 32);
        const float bias_w = (col < 16) ? w0 : w1;
        s[j][r] = sa[r] * 0.125f + bias_h + bias_w;
      }
    }

    // Softmax over 256 keys (rows 0-7 live in lanes 0-15, 8-15 in 16-31).
    float inv[8];
#pragma unroll
    for (int r = 0; r < 8; ++r) {
      float m = s[0][r];
#pragma unroll
      for (int j = 1; j < 16; ++j) m = fmaxf(m, s[j][r]);
#pragma unroll
      for (int d = 1; d <= 8; d <<= 1) m = fmaxf(m, __shfl_xor(m, d, 32));
      float sum = 0.f;
#pragma unroll
      for (int j = 0; j < 16; ++j) {
        const float e = __expf(s[j][r] - m);
        s[j][r] = e;
        sum += e;
      }
#pragma unroll
      for (int d = 1; d <= 8; d <<= 1) sum += __shfl_xor(sum, d, 32);
      inv[r] = 1.0f / sum;
    }

    // Write probabilities (bf16) to this wave's LDS tile (same-wave DS order).
#pragma unroll
    for (int j = 0; j < 16; ++j)
#pragma unroll
      for (int r = 0; r < 8; ++r)
        Pw[(r + hf * 8) * 256 + j * 16 + nn] = f2bf(s[j][r] * inv[r]);

    // out = P (16x256) @ V (256x64): 4 channel tiles x 8 K-steps.
#pragma unroll
    for (int dt = 0; dt < 4; ++dt) {
      v8f oa = V8F_ZERO;
#pragma unroll
      for (int kb = 0; kb < 8; ++kb) {
        const Frag ap = load_afrag(Pw, nn, 256, kb * 32, hf);
        const Frag bv = load_bfrag(Vsm, dt * 16 + nn, 256, kb * 32, hf);
        oa = wmma_bf16(ap, bv, oa);
      }
#pragma unroll
      for (int r = 0; r < 8; ++r) {
        const int tok = rblk * 16 + r + hf * 8;
        Ob[((size_t)bidx * 256 + tok) * 768 + head * 64 + dt * 16 + nn] =
            f2bf(oa[r]);
      }
    }
  }
}

// ---------------------------------------------------------------------------
// Kernel 3: proj GEMM.  out[65536, 768] = O_bf16 @ proj_w + proj_b (fp32 out).
// A tile staged via async global->LDS DMA (pure bf16 copy).
// ---------------------------------------------------------------------------
__global__ __launch_bounds__(256) void proj_gemm_kernel(
    const unsigned short* __restrict__ A, const float* __restrict__ w,
    const float* __restrict__ bias, float* __restrict__ out) {
  __shared__ unsigned short As[128 * 32];
  __shared__ unsigned short Bt[64 * 32];

  const int tid  = threadIdx.x;
  const int wv   = tid >> 5;
  const int lane = tid & 31;
  const int hf   = lane >> 4;
  const int nn   = lane & 15;

  const int m0 = blockIdx.y * 128;
  const int n0 = blockIdx.x * 64;

  v8f acc[4] = {V8F_ZERO, V8F_ZERO, V8F_ZERO, V8F_ZERO};

  for (int ks = 0; ks < 24; ++ks) {
    const int k0 = ks * 32;
    // Stage A (bf16 copy): async DMA, 512 x 16B, 2 per thread.
#pragma unroll
    for (int i = 0; i < 2; ++i) {
      const int idx = tid + i * 256;
      const int row = idx >> 2, c8 = idx & 3;
      const unsigned voff =
          (unsigned)(((size_t)(m0 + row) * 768 + k0 + c8 * 8) * 2u);
      async_copy_b128(A, voff, As + row * 32 + c8 * 8);
    }
    // Stage B transposed (fp32 -> bf16): 512 float4, 2 per thread.
#pragma unroll
    for (int i = 0; i < 2; ++i) {
      const int idx = tid + i * 256;
      const int kk = idx >> 4, n4 = idx & 15;
      const float* src = w + (size_t)(k0 + kk) * 768 + n0 + n4 * 4;
      const float4 vv = *(const float4*)(src);
      __builtin_prefetch(src + 32, 0, 1);   // 32 columns ahead in the same W row
      Bt[(n4 * 4 + 0) * 32 + kk] = f2bf(vv.x);
      Bt[(n4 * 4 + 1) * 32 + kk] = f2bf(vv.y);
      Bt[(n4 * 4 + 2) * 32 + kk] = f2bf(vv.z);
      Bt[(n4 * 4 + 3) * 32 + kk] = f2bf(vv.w);
    }
    wait_asynccnt0();
    __syncthreads();
    const Frag a = load_afrag(As, wv * 16 + nn, 32, 0, hf);
#pragma unroll
    for (int j = 0; j < 4; ++j) {
      const Frag b = load_bfrag(Bt, j * 16 + nn, 32, 0, hf);
      acc[j] = wmma_bf16(a, b, acc[j]);
    }
    __syncthreads();
  }

#pragma unroll
  for (int j = 0; j < 4; ++j) {
    const int n_out = n0 + j * 16 + nn;
    const float bv  = bias[n_out];
#pragma unroll
    for (int r = 0; r < 8; ++r) {
      const int row = m0 + wv * 16 + r + hf * 8;
      out[(size_t)row * 768 + n_out] = acc[j][r] + bv;
    }
  }
}

// ---------------------------------------------------------------------------
extern "C" void kernel_launch(void* const* d_in, const int* in_sizes, int n_in,
                              void* d_out, int out_size, void* d_ws,
                              size_t ws_size, hipStream_t stream) {
  (void)in_sizes; (void)n_in; (void)out_size; (void)ws_size;
  const float* x     = (const float*)d_in[0];
  const float* qkvw  = (const float*)d_in[1];
  const float* qkvb  = (const float*)d_in[2];
  const float* projw = (const float*)d_in[3];
  const float* projb = (const float*)d_in[4];
  const float* rph   = (const float*)d_in[5];
  const float* rpw   = (const float*)d_in[6];

  // Workspace: Q, K (row-major per head), V^T, attention output (all bf16).
  const size_t per = (size_t)256 * 12 * 256 * 64;   // 50,331,648 elems each
  unsigned short* Qb = (unsigned short*)d_ws;
  unsigned short* Kb = Qb + per;
  unsigned short* Vt = Kb + per;
  unsigned short* Ob = Vt + per;

  qkv_gemm_kernel<<<dim3(36, 512), 256, 0, stream>>>(x, qkvw, qkvb, Qb, Kb, Vt);

  const int smem =
      (256 * 64 + 64 * 256 + 32 * 64 + 32 * 64 + 8 * 16 * 256) * 2;  // 139264 B
  hipFuncSetAttribute(reinterpret_cast<const void*>(&attn_kernel),
                      hipFuncAttributeMaxDynamicSharedMemorySize, smem);
  attn_kernel<<<3072, 256, smem, stream>>>(Qb, Kb, Vt, rph, rpw, Ob);

  proj_gemm_kernel<<<dim3(12, 512), 256, 0, stream>>>(Ob, projw, projb,
                                                      (float*)d_out);
}
